// TFS_Head_59674275610646
// MI455X (gfx1250) — matrix operand order repaired and checked by reading, not compiled, not run on hardware
//
#include <hip/hip_runtime.h>
#include <hip/hip_bf16.h>
#include <math.h>

// ---------------------------------------------------------------------------
// Types for CDNA5 WMMA (wave32): A/B = v16h (16 x f16), C/D = v8f (8 x f32)
// ---------------------------------------------------------------------------
typedef __attribute__((ext_vector_type(16))) _Float16 v16h;
typedef __attribute__((ext_vector_type(8)))  _Float16 h8;
typedef __attribute__((ext_vector_type(4)))  _Float16 h4;
typedef __attribute__((ext_vector_type(8)))  float    v8f;
typedef __attribute__((ext_vector_type(4)))  float    f4;

#define H8CAT(lo, hi) __builtin_shufflevector((lo), (hi), 0,1,2,3,4,5,6,7,8,9,10,11,12,13,14,15)

static __device__ __forceinline__ int iminc(int a, int b) { return a < b ? a : b; }

// ---------------------------------------------------------------------------
// Generic batched GEMM:  C = epi( A[M,K] * W[K,N] (+bias) (+residual) )
//   wlay=0 : W element (k,n) at W[k*ldw + n]      (row-major [K,N])
//   wlay=1 : W element (k,n) at W[n*ldw + k]      (i.e. multiply by W^T)
//   epi bit0: +bias[n]   bit1: exact GELU   bit2: +R[m*ldc+n] (residual)
// Batched over grid.z: z -> (zb = z/H2, zh = z%H2), per-axis strides.
// Tile: 64x64 per block (256 thr = 8 waves), each wave 16(M) x 32(N),
// K-step 32 matching v_wmma_f32_16x16x32_f16. fp32 -> f16 conversion in LDS.
// Fast path (interior tiles): unguarded float4 loads + double-buffered LDS.
// Edge path: branchless clamped loads (no exec-mask serialization).
// ---------------------------------------------------------------------------
struct GP {
  const float* A; const float* W; const float* bias; const float* R; float* C;
  int M, N, K;
  int lda, ldw, ldc;
  int wlay, epi, H2;
  long sAb, sAh, sWb, sWh, sCb, sCh;
};

__global__ __launch_bounds__(256) void gemm_wmma_kernel(GP p)
{
  __shared__ alignas(16) _Float16 As[2][64][40];   // [buf][m][k], padded rows
  __shared__ alignas(16) _Float16 Bt[2][64][40];   // [buf][n][k] (W transposed)

  const int tid  = threadIdx.x;
  const int wid  = tid >> 5;          // 8 waves (wave32)
  const int lane = tid & 31;
  const int wm   = wid >> 1;          // 0..3  -> M sub-tile
  const int wn   = wid & 1;           // 0..1  -> N sub-tile (32 wide)
  const int m16  = lane & 15;
  const int kh   = lane >> 4;         // lane half selects K group

  const int bm = blockIdx.y * 64;
  const int bn = blockIdx.x * 64;
  const int z  = blockIdx.z;
  const int zb = z / p.H2, zh = z % p.H2;

  const float* __restrict__ A = p.A + zb * p.sAb + zh * p.sAh;
  const float* __restrict__ W = p.W + zb * p.sWb + zh * p.sWh;
  float*       __restrict__ C = p.C + zb * p.sCb + zh * p.sCh;
  const float* __restrict__ R = p.R ? (p.R + zb * p.sCb + zh * p.sCh) : nullptr;

  v8f acc0 = {}; v8f acc1 = {};

  const bool full = (bm + 64 <= p.M) && (bn + 64 <= p.N) &&
                    ((p.K & 31) == 0) && ((p.lda & 3) == 0);

  if (full) {
    // ---------------- fast path: unguarded, double buffered ----------------
    const bool vecW = ((p.ldw & 3) == 0);
    const int  nkt  = p.K >> 5;

    // per-thread fixed tile coordinates
    const int ar = tid >> 3;            // A rows: ar, ar+32
    const int ac = (tid & 7) * 4;       // A cols: ac..ac+3
    f4 ra0, ra1;                        // A staging regs
    f4 rwv0, rwv1;                      // W staging regs (vector cases)
    float rws[8];                       // W staging regs (scalar case)

    auto loadTiles = [&](int k0) {
      const float* Ab = A + k0 + ac;
      ra0 = *(const f4*)(Ab + (long)(bm + ar)      * p.lda);
      ra1 = *(const f4*)(Ab + (long)(bm + ar + 32) * p.lda);
      if (p.wlay == 1) {
        // W^T rows: n = (tid>>3)+{0,32}, k cols contiguous
        const float* Wb = W + k0 + ac;
        rwv0 = *(const f4*)(Wb + (long)(bn + ar)      * p.ldw);
        rwv1 = *(const f4*)(Wb + (long)(bn + ar + 32) * p.ldw);
      } else if (vecW) {
        // W rows k = (tid>>4)+{0,16}, n = (tid&15)*4 contiguous
        const float* Wb = W + bn + (tid & 15) * 4;
        rwv0 = *(const f4*)(Wb + (long)(k0 + (tid >> 4))      * p.ldw);
        rwv1 = *(const f4*)(Wb + (long)(k0 + 16 + (tid >> 4)) * p.ldw);
      } else {
        // odd ldw (vocab matrix): 8 unguarded scalar loads, still pipelined
        const float* Wb = W + bn + (tid & 63);
        const int kb = tid >> 6;
        #pragma unroll
        for (int e = 0; e < 8; ++e)
          rws[e] = Wb[(long)(k0 + kb + e * 4) * p.ldw];
      }
    };

    auto storeTiles = [&](int buf) {
      h4 ha0, ha1;
      #pragma unroll
      for (int j = 0; j < 4; ++j) { ha0[j] = (_Float16)ra0[j]; ha1[j] = (_Float16)ra1[j]; }
      *(h4*)&As[buf][ar][ac]      = ha0;
      *(h4*)&As[buf][ar + 32][ac] = ha1;
      if (p.wlay == 1) {
        h4 hw0, hw1;
        #pragma unroll
        for (int j = 0; j < 4; ++j) { hw0[j] = (_Float16)rwv0[j]; hw1[j] = (_Float16)rwv1[j]; }
        *(h4*)&Bt[buf][ar][ac]      = hw0;
        *(h4*)&Bt[buf][ar + 32][ac] = hw1;
      } else if (vecW) {
        const int k = tid >> 4, n = (tid & 15) * 4;
        #pragma unroll
        for (int j = 0; j < 4; ++j) {
          Bt[buf][n + j][k]      = (_Float16)rwv0[j];
          Bt[buf][n + j][k + 16] = (_Float16)rwv1[j];
        }
      } else {
        const int n = tid & 63, kb = tid >> 6;
        #pragma unroll
        for (int e = 0; e < 8; ++e)
          Bt[buf][n][kb + e * 4] = (_Float16)rws[e];
      }
    };

    loadTiles(0);
    storeTiles(0);
    __syncthreads();

    int buf = 0;
    for (int kt = 0; kt < nkt; ++kt) {
      const bool more = (kt + 1 < nkt);
      if (more) loadTiles((kt + 1) << 5);   // issue next tile's global loads

      // ---- fragments, contiguous k = kh*16 + s (read current buffer) ----
      h8 alo = *(const h8*)&As[buf][wm * 16 + m16][kh * 16];
      h8 ahi = *(const h8*)&As[buf][wm * 16 + m16][kh * 16 + 8];
      v16h av = H8CAT(alo, ahi);
      h8 b0l = *(const h8*)&Bt[buf][wn * 32 + m16][kh * 16];
      h8 b0h = *(const h8*)&Bt[buf][wn * 32 + m16][kh * 16 + 8];
      v16h bv0 = H8CAT(b0l, b0h);
      h8 b1l = *(const h8*)&Bt[buf][wn * 32 + 16 + m16][kh * 16];
      h8 b1h = *(const h8*)&Bt[buf][wn * 32 + 16 + m16][kh * 16 + 8];
      v16h bv1 = H8CAT(b1l, b1h);

      acc0 = __builtin_amdgcn_wmma_f32_16x16x32_f16(false, av, false, bv0,
                                                    (short)0, acc0, false, false);
      acc1 = __builtin_amdgcn_wmma_f32_16x16x32_f16(false, av, false, bv1,
                                                    (short)0, acc1, false, false);
      __syncthreads();
      if (more) {
        storeTiles(buf ^ 1);
        buf ^= 1;
        __syncthreads();
      }
    }
  } else {
    // -------- edge path: branchless clamped loads (always in-bounds) -------
    const int Mc = p.M - 1, Kc = p.K - 1, Nc = p.N - 1;
    for (int k0 = 0; k0 < p.K; k0 += 32) {
      float va[8]; bool oa[8];
      #pragma unroll
      for (int e = 0; e < 8; ++e) {
        int li = e * 256 + tid;
        int r = li >> 5, c = li & 31;
        int gm = bm + r, gk = k0 + c;
        oa[e] = (gm < p.M) && (gk < p.K);
        va[e] = A[(long)iminc(gm, Mc) * p.lda + iminc(gk, Kc)];
      }
      float vw[8]; bool ow[8];
      if (p.wlay == 0) {
        #pragma unroll
        for (int e = 0; e < 8; ++e) {
          int li = e * 256 + tid;
          int n = li & 63, k = li >> 6;
          int gn = bn + n, gk = k0 + k;
          ow[e] = (gn < p.N) && (gk < p.K);
          vw[e] = W[(long)iminc(gk, Kc) * p.ldw + iminc(gn, Nc)];
        }
      } else {
        #pragma unroll
        for (int e = 0; e < 8; ++e) {
          int li = e * 256 + tid;
          int k = li & 31, n = li >> 5;
          int gn = bn + n, gk = k0 + k;
          ow[e] = (gn < p.N) && (gk < p.K);
          vw[e] = W[(long)iminc(gn, Nc) * p.ldw + iminc(gk, Kc)];
        }
      }
      #pragma unroll
      for (int e = 0; e < 8; ++e) {
        int li = e * 256 + tid;
        int r = li >> 5, c = li & 31;
        As[0][r][c] = (_Float16)(oa[e] ? va[e] : 0.f);
      }
      if (p.wlay == 0) {
        #pragma unroll
        for (int e = 0; e < 8; ++e) {
          int li = e * 256 + tid;
          int n = li & 63, k = li >> 6;
          Bt[0][n][k] = (_Float16)(ow[e] ? vw[e] : 0.f);
        }
      } else {
        #pragma unroll
        for (int e = 0; e < 8; ++e) {
          int li = e * 256 + tid;
          int k = li & 31, n = li >> 5;
          Bt[0][n][k] = (_Float16)(ow[e] ? vw[e] : 0.f);
        }
      }
      __syncthreads();

      h8 alo = *(const h8*)&As[0][wm * 16 + m16][kh * 16];
      h8 ahi = *(const h8*)&As[0][wm * 16 + m16][kh * 16 + 8];
      v16h av = H8CAT(alo, ahi);
      h8 b0l = *(const h8*)&Bt[0][wn * 32 + m16][kh * 16];
      h8 b0h = *(const h8*)&Bt[0][wn * 32 + m16][kh * 16 + 8];
      v16h bv0 = H8CAT(b0l, b0h);
      h8 b1l = *(const h8*)&Bt[0][wn * 32 + 16 + m16][kh * 16];
      h8 b1h = *(const h8*)&Bt[0][wn * 32 + 16 + m16][kh * 16 + 8];
      v16h bv1 = H8CAT(b1l, b1h);

      acc0 = __builtin_amdgcn_wmma_f32_16x16x32_f16(false, av, false, bv0,
                                                    (short)0, acc0, false, false);
      acc1 = __builtin_amdgcn_wmma_f32_16x16x32_f16(false, av, false, bv1,
                                                    (short)0, acc1, false, false);
      __syncthreads();
    }
  }

  // ---- epilogue: C/D layout VGPR v -> M = wm*16 + kh*8 + v ; N = lane&15 ----
  #pragma unroll
  for (int v = 0; v < 8; ++v) {
    int m  = bm + wm * 16 + kh * 8 + v;
    int n0 = bn + wn * 32 + m16;
    float vals[2] = { acc0[v], acc1[v] };
    #pragma unroll
    for (int j = 0; j < 2; ++j) {
      int n = n0 + j * 16;
      if (m < p.M && n < p.N) {
        float x = vals[j];
        if (p.epi & 1) x += p.bias[n];
        if (p.epi & 2) x = 0.5f * x * (1.f + erff(x * 0.70710678118654752f));
        if (p.epi & 4) x += R[(long)m * p.ldc + n];
        C[(long)m * p.ldc + n] = x;
      }
    }
  }
}

// ---------------------------------------------------------------------------
// Embedding: h[b,s,:] = wte[ids[b,s],:] + wpe[s,:]
// ---------------------------------------------------------------------------
__global__ void embed_kernel(const int* __restrict__ ids,
                             const float* __restrict__ wte,
                             const float* __restrict__ wpe,
                             float* __restrict__ h, int S, int Dm)
{
  int bs = blockIdx.x;
  int s  = bs % S;
  int id = ids[bs];
  long ho = (long)bs * Dm;
  for (int d = threadIdx.x; d < Dm; d += 256)
    h[ho + d] = wte[(long)id * Dm + d] + wpe[(long)s * Dm + d];
}

// ---------------------------------------------------------------------------
// ctx = concat(img+img_pos, txt+txt_pos, klg+klg_pos) along seq -> [B,160,D]
// ---------------------------------------------------------------------------
__global__ void ctx_kernel(const float* __restrict__ img,
                           const float* __restrict__ txt,
                           const float* __restrict__ klg,
                           const float* __restrict__ ip,
                           const float* __restrict__ tp,
                           const float* __restrict__ kp,
                           float* __restrict__ ctx, int Dm)
{
  int bj = blockIdx.x;
  int b = bj / 160, j = bj % 160;
  const float* src; const float* pos;
  if (j < 80)       { src = img + ((long)b * 80 + j)        * Dm; pos = ip + (long)j        * Dm; }
  else if (j < 120) { src = txt + ((long)b * 40 + (j - 80)) * Dm; pos = tp + (long)(j - 80) * Dm; }
  else              { src = klg + ((long)b * 40 + (j - 120))* Dm; pos = kp + (long)(j - 120)* Dm; }
  long o = (long)bj * Dm;
  for (int d = threadIdx.x; d < Dm; d += 256) ctx[o + d] = src[d] + pos[d];
}

// ---------------------------------------------------------------------------
// Off-by-one softmax over rows of nk logits (in place), wave-per-row (wave32).
// x' = scale*x (+ causal mask);  p = exp(x'-max) / (1 + sum exp(x'-max))
// ---------------------------------------------------------------------------
__global__ void softmax1_kernel(float* __restrict__ x, int rows, int nk, int nq,
                                float scale, int causal)
{
  int wid = threadIdx.x >> 5, lane = threadIdx.x & 31;
  int row = blockIdx.x * 8 + wid;
  if (row >= rows) return;
  int q = row % nq;
  float* p = x + (long)row * nk;

  float mx = -3.4e38f;
  for (int k = lane; k < nk; k += 32) {
    float v = p[k] * scale + ((causal && k > q) ? -1000000.0f : 0.f);
    mx = fmaxf(mx, v);
  }
  for (int o = 16; o > 0; o >>= 1) mx = fmaxf(mx, __shfl_xor(mx, o, 32));

  float s = 0.f;
  for (int k = lane; k < nk; k += 32) {
    float v = p[k] * scale + ((causal && k > q) ? -1000000.0f : 0.f);
    s += __expf(v - mx);
  }
  for (int o = 16; o > 0; o >>= 1) s += __shfl_xor(s, o, 32);

  float inv = 1.f / (1.f + s);
  for (int k = lane; k < nk; k += 32) {
    float v = p[k] * scale + ((causal && k > q) ? -1000000.0f : 0.f);
    p[k] = __expf(v - mx) * inv;
  }
}

// ---------------------------------------------------------------------------
// LayerNorm over last dim Dw, one block per row.
// ---------------------------------------------------------------------------
__global__ void ln_kernel(const float* __restrict__ x,
                          const float* __restrict__ g,
                          const float* __restrict__ b,
                          float* __restrict__ y, int Dw)
{
  __shared__ float red[256];
  long base = (long)blockIdx.x * Dw;

  float s = 0.f;
  for (int d = threadIdx.x; d < Dw; d += 256) s += x[base + d];
  red[threadIdx.x] = s; __syncthreads();
  for (int o = 128; o > 0; o >>= 1) {
    if (threadIdx.x < o) red[threadIdx.x] += red[threadIdx.x + o];
    __syncthreads();
  }
  float mu = red[0] / (float)Dw;
  __syncthreads();

  float v = 0.f;
  for (int d = threadIdx.x; d < Dw; d += 256) {
    float t = x[base + d] - mu; v += t * t;
  }
  red[threadIdx.x] = v; __syncthreads();
  for (int o = 128; o > 0; o >>= 1) {
    if (threadIdx.x < o) red[threadIdx.x] += red[threadIdx.x + o];
    __syncthreads();
  }
  float rs = rsqrtf(red[0] / (float)Dw + 1e-5f);

  for (int d = threadIdx.x; d < Dw; d += 256)
    y[base + d] = (x[base + d] - mu) * rs * g[d] + b[d];
}

// ---------------------------------------------------------------------------
// Host side
// ---------------------------------------------------------------------------
static inline void G(hipStream_t st,
                     const float* A, int lda,
                     const float* W, int ldw, int wlay,
                     const float* bias, const float* R,
                     float* C, int ldc,
                     int M, int N, int K, int epi,
                     int nb = 1, int H2 = 1,
                     long sAb = 0, long sAh = 0, long sWb = 0, long sWh = 0,
                     long sCb = 0, long sCh = 0)
{
  GP p;
  p.A = A; p.W = W; p.bias = bias; p.R = R; p.C = C;
  p.M = M; p.N = N; p.K = K;
  p.lda = lda; p.ldw = ldw; p.ldc = ldc;
  p.wlay = wlay; p.epi = epi; p.H2 = H2;
  p.sAb = sAb; p.sAh = sAh; p.sWb = sWb; p.sWh = sWh; p.sCb = sCb; p.sCh = sCh;
  dim3 grid((N + 63) / 64, (M + 63) / 64, nb);
  gemm_wmma_kernel<<<grid, 256, 0, st>>>(p);
}

extern "C" void kernel_launch(void* const* d_in, const int* in_sizes, int n_in,
                              void* d_out, int out_size, void* d_ws, size_t ws_size,
                              hipStream_t stream)
{
  (void)in_sizes; (void)n_in; (void)out_size; (void)ws_size;
  const int B = 8, S = 40, Dm = 1024, Hh = 8, DH = 1024, HD = 8192;
  const int Vv = 50257, Ll = 2, NC = 160;
  const int Mt = B * S;         // 320 decode tokens
  const int Mc = B * NC;        // 1280 ctx tokens
  const float scale = 0.03125f; // 1/sqrt(1024)

  // ---- inputs (setup_inputs dict order, params flattened in insertion order)
  const int*   ids  = (const int*)  d_in[0];
  const float* img  = (const float*)d_in[1];
  const float* txt  = (const float*)d_in[2];
  const float* klg  = (const float*)d_in[3];
  const float* wte  = (const float*)d_in[4];
  const float* wpe  = (const float*)d_in[5];
  const float* ipos = (const float*)d_in[6];
  const float* tpos = (const float*)d_in[7];
  const float* kpos = (const float*)d_in[8];
  const float* cWq = (const float*)d_in[9],  *cWk = (const float*)d_in[10];
  const float* cWv = (const float*)d_in[11], *cWo = (const float*)d_in[12];
  const float* cbq = (const float*)d_in[13], *cbk = (const float*)d_in[14];
  const float* cbv = (const float*)d_in[15], *cbo = (const float*)d_in[16];
  const float* sWq = (const float*)d_in[17], *sWk = (const float*)d_in[18];
  const float* sWv = (const float*)d_in[19], *sWo = (const float*)d_in[20];
  const float* sbq = (const float*)d_in[21], *sbk = (const float*)d_in[22];
  const float* sbv = (const float*)d_in[23], *sbo = (const float*)d_in[24];
  const float* ln1g = (const float*)d_in[25], *ln1b = (const float*)d_in[26];
  const float* ln2g = (const float*)d_in[27], *ln2b = (const float*)d_in[28];
  const float* fW1 = (const float*)d_in[29], *fb1 = (const float*)d_in[30];
  const float* fW2 = (const float*)d_in[31], *fb2 = (const float*)d_in[32];
  const float* oW1 = (const float*)d_in[33], *ob1 = (const float*)d_in[34];
  const float* olng = (const float*)d_in[35], *olnb = (const float*)d_in[36];
  const float* oW2 = (const float*)d_in[37], *ob2 = (const float*)d_in[38];
  float* out = (float*)d_out;

  // ---- workspace carve-up (fp32) ----
  float* ws   = (float*)d_ws;
  float* x    = ws; ws += (long)Mt * Dm;       // hidden state [320,1024]
  float* ctx  = ws; ws += (long)Mc * Dm;       // [1280,1024]
  float* nbuf = ws; ws += (long)Mt * Dm;       // LN output
  float* qb   = ws; ws += (long)Mt * HD;       // [320,8192]
  float* kb   = ws; ws += (long)Mc * HD;       // [1280,8192]
  float* vb   = ws; ws += (long)Mc * HD;
  float* att  = ws; ws += (long)Mt * HD;       // attn output pre-Wo
  float* lg   = ws; ws += (long)B * Hh * S * NC; // logits/probs (max 64*40*160)
  float* mid  = ws; ws += (long)Mt * 2 * Dm;   // FFN mid / head mid [320,2048]
  float* mid2 = ws; ws += (long)Mt * 2 * Dm;   // LN'd head mid

  // ---- embeddings + context ----
  embed_kernel<<<Mt, 256, 0, stream>>>(ids, wte, wpe, x, S, Dm);
  ctx_kernel<<<Mc, 256, 0, stream>>>(img, txt, klg, ipos, tpos, kpos, ctx, Dm);

  for (int i = 0; i < Ll; ++i) {
    const long wO = (long)i * Dm * HD;              // [D,HD] weight offset
    const long oO = (long)i * HD * Dm;              // [HD,D] weight offset
    // ================= cross attention (kv = ctx, no mask) =================
    G(stream, x,   Dm, cWq + wO, HD, 0, cbq + (long)i * HD, nullptr, qb, HD, Mt, HD, Dm, 1);
    G(stream, ctx, Dm, cWk + wO, HD, 0, cbk + (long)i * HD, nullptr, kb, HD, Mc, HD, Dm, 1);
    G(stream, ctx, Dm, cWv + wO, HD, 0, cbv + (long)i * HD, nullptr, vb, HD, Mc, HD, Dm, 1);
    // logits[b,h] = q[b,:,h*DH:+DH] @ k[b,:,h*DH:+DH]^T  -> [40,160]
    G(stream, qb, HD, kb, HD, 1, nullptr, nullptr, lg, NC, S, NC, DH, 0,
      B * Hh, Hh, (long)S * HD, DH, (long)NC * HD, DH, (long)Hh * S * NC, (long)S * NC);
    softmax1_kernel<<<(B * Hh * S + 7) / 8, 256, 0, stream>>>(lg, B * Hh * S, NC, S, scale, 0);
    // att[b,:,h*DH:+DH] = probs @ v[b,:,h*DH:+DH]
    G(stream, lg, NC, vb, HD, 0, nullptr, nullptr, att, HD, S, DH, NC, 0,
      B * Hh, Hh, (long)Hh * S * NC, (long)S * NC, (long)NC * HD, DH, (long)S * HD, DH);
    // x = x + att @ cWo + cbo
    G(stream, att, HD, cWo + oO, Dm, 0, cbo + (long)i * Dm, x, x, Dm, Mt, Dm, HD, 1 | 4);

    // ================= self attention (pre-LN, causal mask) ================
    ln_kernel<<<Mt, 256, 0, stream>>>(x, ln1g + (long)i * Dm, ln1b + (long)i * Dm, nbuf, Dm);
    G(stream, nbuf, Dm, sWq + wO, HD, 0, sbq + (long)i * HD, nullptr, qb, HD, Mt, HD, Dm, 1);
    G(stream, nbuf, Dm, sWk + wO, HD, 0, sbk + (long)i * HD, nullptr, kb, HD, Mt, HD, Dm, 1);
    G(stream, nbuf, Dm, sWv + wO, HD, 0, sbv + (long)i * HD, nullptr, vb, HD, Mt, HD, Dm, 1);
    G(stream, qb, HD, kb, HD, 1, nullptr, nullptr, lg, S, S, S, DH, 0,
      B * Hh, Hh, (long)S * HD, DH, (long)S * HD, DH, (long)Hh * S * S, (long)S * S);
    softmax1_kernel<<<(B * Hh * S + 7) / 8, 256, 0, stream>>>(lg, B * Hh * S, S, S, scale, 1);
    G(stream, lg, S, vb, HD, 0, nullptr, nullptr, att, HD, S, DH, S, 0,
      B * Hh, Hh, (long)Hh * S * S, (long)S * S, (long)S * HD, DH, (long)S * HD, DH);
    G(stream, att, HD, sWo + oO, Dm, 0, sbo + (long)i * Dm, x, x, Dm, Mt, Dm, HD, 1 | 4);

    // ================= FFN: x = x + GELU(LN(x)W1+b1)W2+b2 ==================
    ln_kernel<<<Mt, 256, 0, stream>>>(x, ln2g + (long)i * Dm, ln2b + (long)i * Dm, nbuf, Dm);
    G(stream, nbuf, Dm, fW1 + (long)i * Dm * Dm, Dm, 0, fb1 + (long)i * Dm,
      nullptr, mid, Dm, Mt, Dm, Dm, 1 | 2);
    G(stream, mid, Dm, fW2 + (long)i * Dm * Dm, Dm, 0, fb2 + (long)i * Dm,
      x, x, Dm, Mt, Dm, Dm, 1 | 4);
  }

  // ===== head: Linear -> GELU -> LayerNorm -> Linear(vocab) =====
  G(stream, x, Dm, oW1, 2 * Dm, 0, ob1, nullptr, mid, 2 * Dm, Mt, 2 * Dm, Dm, 1 | 2);
  ln_kernel<<<Mt, 256, 0, stream>>>(mid, olng, olnb, mid2, 2 * Dm);
  G(stream, mid2, 2 * Dm, oW2, Vv, 0, ob2, nullptr, out, Vv, Mt, Vv, 2 * Dm, 1);
}
